// LSTMModel_34033320853539
// MI455X (gfx1250) — matrix operand order repaired and checked by reading, not compiled, not run on hardware
//
#include <hip/hip_runtime.h>
#include <hip/hip_bf16.h>

typedef _Float16 v16h __attribute__((ext_vector_type(16)));
typedef _Float16 v8h  __attribute__((ext_vector_type(8)));
typedef float    v8f  __attribute__((ext_vector_type(8)));

#define GLB __attribute__((address_space(1)))

#define BATCH 256
#define TSTEPS 512
#define DIN 128
#define HID 256
#define MWG 16          // batch rows per workgroup
#define NT_TOTAL 64     // 4H/16 n-tiles
#define KT0 12          // (128+256)/32 k-tiles, layer 0
#define KT1 16          // (256+256)/32 k-tiles, layer 1

// ---------------- workspace layout (bytes) ----------------
#define WS_W1   786432     // w0 frags: 64*12*512 halves = 786432 B at [0)
#define WS_B0   1835008    // w1 frags: 64*16*512 halves = 1048576 B
#define WS_B1   1839104
#define WS_LAST 1843200    // last_h2: 256*256 f32; total ~2.01 MB

// ---------------------------------------------------------------------------
// Pack weights into per-WMMA B-fragment layout (f16).
// B[k][n] = Wcat[n][k] (gates = in @ Wcat^T). Fragment (ntile,ktile):
// lane l: n = ntile*16 + (l&15); half e: k = ktile*32 + (l>=16 ? 8:0) + (e&7) + (e>=8 ? 16:0).
// Fragment stored as 32 lanes x 16 contiguous halves (1 KB).
// ---------------------------------------------------------------------------
__global__ void pack_weights_kernel(const float* __restrict__ Wih0,
                                    const float* __restrict__ Whh0,
                                    const float* __restrict__ Wih1,
                                    const float* __restrict__ Whh1,
                                    _Float16* __restrict__ w0f,
                                    _Float16* __restrict__ w1f) {
  int id = blockIdx.x * 256 + threadIdx.x;
  const int L0 = NT_TOTAL * KT0 * 512;   // 393216 halves in layer-0 region
  int frag, r, kt_cnt;
  _Float16* dst;
  bool layer0 = (id < L0);
  if (layer0) { frag = id >> 9; r = id & 511; kt_cnt = KT0; dst = w0f; }
  else { int id1 = id - L0; frag = id1 >> 9; r = id1 & 511; kt_cnt = KT1; dst = w1f; }
  int ntile = frag / kt_cnt;
  int ktile = frag % kt_cnt;
  int lane = r >> 4;
  int e = r & 15;
  int n = ntile * 16 + (lane & 15);
  int k = ktile * 32 + ((lane >= 16) ? 8 : 0) + (e & 7) + ((e >> 3) << 4);
  float v;
  if (layer0) {
    v = (k < DIN) ? Wih0[n * DIN + k] : Whh0[n * HID + (k - DIN)];
  } else {
    v = (k < HID) ? Wih1[n * HID + k] : Whh1[n * HID + (k - HID)];
  }
  dst[(size_t)frag * 512 + r] = (_Float16)v;
}

__global__ void pack_bias_kernel(const float* bih0, const float* bhh0,
                                 const float* bih1, const float* bhh1,
                                 float* bias0, float* bias1) {
  int id = blockIdx.x * 256 + threadIdx.x;
  if (id < 1024) bias0[id] = bih0[id] + bhh0[id];
  else { int j = id - 1024; bias1[j] = bih1[j] + bhh1[j]; }
}

// ---------------------------------------------------------------------------
// A fragment (16x32 f16, ISA layout: lane m holds K koff..koff+7 and
// 16+koff..16+koff+7, koff = lane<16 ? 0 : 8) from row-major LDS [16][ldk].
// ---------------------------------------------------------------------------
__device__ __forceinline__ v16h load_a_frag(const _Float16* base, int ldk,
                                            int kt, int m_lo, int koff) {
  const _Float16* p = base + m_lo * ldk + kt * 32 + koff;
  v8h lo = *(const v8h*)(p);
  v8h hi = *(const v8h*)(p + 16);
  return __builtin_shufflevector(lo, hi, 0,1,2,3,4,5,6,7,8,9,10,11,12,13,14,15);
}

// B fragment: explicit global (address_space(1)) load -> global_load_b128 x2,
// LOADcnt-only tracking (keeps the DS pipe free for A-fragment ds_loads).
__device__ __forceinline__ v16h load_b_frag(const GLB _Float16* base, size_t off) {
  return *(const GLB v16h*)(base + off);
}

__device__ __forceinline__ float fsigmoid(float x) {
  return 1.0f / (1.0f + __expf(-x));
}
__device__ __forceinline__ float ftanh(float x) {
  return 2.0f / (1.0f + __expf(-2.0f * x)) - 1.0f;
}

// ---------------------------------------------------------------------------
// Fused 2-layer LSTM. One workgroup per 16 batch rows, persistent over T.
// 8 waves; wave w owns gate columns [32w, 32w+32) of each of i/f/g/o.
// ---------------------------------------------------------------------------
__global__ __launch_bounds__(256, 1)
void lstm_fused_kernel(const float* __restrict__ x,
                       const int* __restrict__ mask,
                       const _Float16* __restrict__ w0f,
                       const _Float16* __restrict__ w1f,
                       const float* __restrict__ bias0,
                       const float* __restrict__ bias1,
                       float* __restrict__ last) {
  __shared__ _Float16 xs[MWG * DIN];     // x_t tile, f16
  __shared__ _Float16 h0s[MWG * HID];    // layer-0 hidden state
  __shared__ _Float16 h1s[MWG * HID];    // layer-1 hidden state
  __shared__ int lens[MWG];

  const int tid = threadIdx.x;
  const int wv = tid >> 5;
  const int lane = tid & 31;
  const int m_lo = lane & 15;
  const int hihalf = lane >> 4;          // 0: rows 0..7 / K-lo, 1: rows 8..15 / K-hi
  const int koff = hihalf * 8;
  const int b0 = blockIdx.x * MWG;

  const GLB float* xg   = (const GLB float*)x;
  GLB float*       lastg = (GLB float*)last;

  for (int i = tid; i < MWG * HID; i += 256) {
    h0s[i] = (_Float16)0.0f;
    h1s[i] = (_Float16)0.0f;
  }
  if (tid < MWG) {
    const GLB int* mrow = (const GLB int*)(mask + (size_t)(b0 + tid) * TSTEPS);
    int s = 0;
    for (int t = 0; t < TSTEPS; ++t) s += mrow[t];
    lens[tid] = s;
  }

  // hoist per-lane biases (constant over time)
  float bi0[2], bf0[2], bg0[2], bo0[2], bi1[2], bf1[2], bg1[2], bo1[2];
#pragma unroll
  for (int tt = 0; tt < 2; ++tt) {
    int colH = wv * 32 + tt * 16 + m_lo;
    bi0[tt] = bias0[0 * HID + colH]; bf0[tt] = bias0[1 * HID + colH];
    bg0[tt] = bias0[2 * HID + colH]; bo0[tt] = bias0[3 * HID + colH];
    bi1[tt] = bias1[0 * HID + colH]; bf1[tt] = bias1[1 * HID + colH];
    bg1[tt] = bias1[2 * HID + colH]; bo1[tt] = bias1[3 * HID + colH];
  }

  float c0[2][8] = {{0}}, c1[2][8] = {{0}};
  const v8f zacc = {0.f,0.f,0.f,0.f,0.f,0.f,0.f,0.f};

  __syncthreads();

  // hoist the capture times (t == len-1) for this lane's 8 rows
  int tcap[8];
#pragma unroll
  for (int r = 0; r < 8; ++r) tcap[r] = lens[r + hihalf * 8] - 1;

  for (int t = 0; t < TSTEPS; ++t) {
    // Launder the weight-fragment base addresses so they are loop-variant
    // (stops LICM from hoisting + spilling the invariant B loads), then cast
    // back to explicit GLOBAL address space so they lower to global_load_b128.
    unsigned long long w0a = (unsigned long long)w0f;
    unsigned long long w1a = (unsigned long long)w1f;
    asm volatile("" : "+s"(w0a), "+s"(w1a));
    const GLB _Float16* w0p = (const GLB _Float16*)w0a;
    const GLB _Float16* w1p = (const GLB _Float16*)w1a;

    // ---- stage x_t (f32 -> f16), 2048 halves, 8 per thread, coalesced ----
    {
      int base = tid * 8;
      int row = base >> 7;
      int col = base & 127;
      const GLB float* src = xg + ((size_t)(b0 + row) * TSTEPS + t) * DIN + col;
#pragma unroll
      for (int j = 0; j < 8; ++j) xs[base + j] = (_Float16)src[j];
    }
    __syncthreads();   // (A) xs ready; prior-iter h1s writes visible

    // ================= layer 0: gates = [x_t | h0] @ W0^T =================
    {
      v8f acc[8];
#pragma unroll
      for (int j = 0; j < 8; ++j) acc[j] = zacc;
      for (int kt = 0; kt < KT0; ++kt) {
        v16h a = (kt < 4) ? load_a_frag(xs, DIN, kt, m_lo, koff)
                          : load_a_frag(h0s, HID, kt - 4, m_lo, koff);
        v16h bfr[8];
#pragma unroll
        for (int j = 0; j < 8; ++j) {           // cluster 16 global_load_b128
          int g = j >> 1, tt = j & 1;
          int ntile = g * 16 + wv * 2 + tt;
          bfr[j] = load_b_frag(w0p, ((size_t)(ntile * KT0 + kt) << 9) + (lane << 4));
        }
#pragma unroll
        for (int j = 0; j < 8; ++j)
          acc[j] = __builtin_amdgcn_wmma_f32_16x16x32_f16(
              false, a, false, bfr[j], (short)0, acc[j], false, false);
      }
      __syncthreads(); // (B) all waves done reading h0s

#pragma unroll
      for (int tt = 0; tt < 2; ++tt) {
        int colH = wv * 32 + tt * 16 + m_lo;
#pragma unroll
        for (int r = 0; r < 8; ++r) {
          float iv = fsigmoid(acc[0 + tt][r] + bi0[tt]);
          float fv = fsigmoid(acc[2 + tt][r] + bf0[tt]);
          float gv = ftanh   (acc[4 + tt][r] + bg0[tt]);
          float ov = fsigmoid(acc[6 + tt][r] + bo0[tt]);
          float c = fv * c0[tt][r] + iv * gv;
          c0[tt][r] = c;
          float h = ov * ftanh(c);
          int m = r + hihalf * 8;
          h0s[m * HID + colH] = (_Float16)h;
        }
      }
    }
    __syncthreads();   // (C) new h0 visible (layer-1 input)

    // ================ layer 1: gates = [h0_t | h1] @ W1^T ================
    {
      v8f acc[8];
#pragma unroll
      for (int j = 0; j < 8; ++j) acc[j] = zacc;
      for (int kt = 0; kt < KT1; ++kt) {
        v16h a = (kt < 8) ? load_a_frag(h0s, HID, kt, m_lo, koff)
                          : load_a_frag(h1s, HID, kt - 8, m_lo, koff);
        v16h bfr[8];
#pragma unroll
        for (int j = 0; j < 8; ++j) {
          int g = j >> 1, tt = j & 1;
          int ntile = g * 16 + wv * 2 + tt;
          bfr[j] = load_b_frag(w1p, ((size_t)(ntile * KT1 + kt) << 9) + (lane << 4));
        }
#pragma unroll
        for (int j = 0; j < 8; ++j)
          acc[j] = __builtin_amdgcn_wmma_f32_16x16x32_f16(
              false, a, false, bfr[j], (short)0, acc[j], false, false);
      }
      __syncthreads(); // (D) all waves done reading h1s

#pragma unroll
      for (int tt = 0; tt < 2; ++tt) {
        int colH = wv * 32 + tt * 16 + m_lo;
#pragma unroll
        for (int r = 0; r < 8; ++r) {
          float iv = fsigmoid(acc[0 + tt][r] + bi1[tt]);
          float fv = fsigmoid(acc[2 + tt][r] + bf1[tt]);
          float gv = ftanh   (acc[4 + tt][r] + bg1[tt]);
          float ov = fsigmoid(acc[6 + tt][r] + bo1[tt]);
          float c = fv * c1[tt][r] + iv * gv;
          c1[tt][r] = c;
          float h = ov * ftanh(c);
          int m = r + hihalf * 8;
          h1s[m * HID + colH] = (_Float16)h;
          if (t == tcap[r])
            lastg[(size_t)(b0 + m) * HID + colH] = h;  // capture last valid h2
        }
      }
    }
    __syncthreads();   // close the step before xs is overwritten
  }
}

// ---------------------------------------------------------------------------
// out[b] = last[b,:] . fc_w + fc_b
// ---------------------------------------------------------------------------
__global__ void fc_kernel(const float* __restrict__ last,
                          const float* __restrict__ fcw,
                          const float* __restrict__ fcb,
                          float* __restrict__ out) {
  int b = blockIdx.x * blockDim.x + threadIdx.x;
  if (b < BATCH) {
    float s = 0.0f;
    for (int j = 0; j < HID; ++j) s += last[(size_t)b * HID + j] * fcw[j];
    out[b] = s + fcb[0];
  }
}

extern "C" void kernel_launch(void* const* d_in, const int* in_sizes, int n_in,
                              void* d_out, int out_size, void* d_ws, size_t ws_size,
                              hipStream_t stream) {
  const float* x    = (const float*)d_in[0];
  const int*   mask = (const int*)d_in[1];
  const float* Wih0 = (const float*)d_in[2];
  const float* Whh0 = (const float*)d_in[3];
  const float* bih0 = (const float*)d_in[4];
  const float* bhh0 = (const float*)d_in[5];
  const float* Wih1 = (const float*)d_in[6];
  const float* Whh1 = (const float*)d_in[7];
  const float* bih1 = (const float*)d_in[8];
  const float* bhh1 = (const float*)d_in[9];
  const float* fcw  = (const float*)d_in[10];
  const float* fcb  = (const float*)d_in[11];

  char* ws = (char*)d_ws;
  _Float16* w0f   = (_Float16*)(ws);
  _Float16* w1f   = (_Float16*)(ws + WS_W1);
  float*    bias0 = (float*)(ws + WS_B0);
  float*    bias1 = (float*)(ws + WS_B1);
  float*    last  = (float*)(ws + WS_LAST);

  pack_weights_kernel<<<3584, 256, 0, stream>>>(Wih0, Whh0, Wih1, Whh1, w0f, w1f);
  pack_bias_kernel<<<8, 256, 0, stream>>>(bih0, bhh0, bih1, bhh1, bias0, bias1);
  lstm_fused_kernel<<<BATCH / MWG, 256, 0, stream>>>(x, mask, w0f, w1f, bias0, bias1, last);
  fc_kernel<<<1, 256, 0, stream>>>(last, fcw, fcb, (float*)d_out);
}
